// Attention_14955076125142
// MI455X (gfx1250) — matrix operand order repaired and checked by reading, not compiled, not run on hardware
//
#include <hip/hip_runtime.h>

// ---------------------------------------------------------------------------
// Problem constants (from reference): N=65536 arcs, F=1024 features.
// ---------------------------------------------------------------------------
static constexpr int FD    = 1024;    // feature dim
static constexpr int NARC  = 65536;   // rows
static constexpr int TM    = 32;      // rows per block in energies kernel
static constexpr int NCH   = 512;     // rows per context partial chunk

typedef __attribute__((ext_vector_type(16))) __bf16 v16bf;
typedef __attribute__((ext_vector_type(8)))  __bf16 v8bf;
typedef __attribute__((ext_vector_type(8)))  float  v8f;

// ---------------------------------------------------------------------------
// f32 -> bf16 weight conversion (done once per launch, tiny)
// ---------------------------------------------------------------------------
__global__ void cvt_f32_bf16(const float* __restrict__ src,
                             __bf16* __restrict__ dst, int n) {
  int i = blockIdx.x * blockDim.x + threadIdx.x;
  if (i < n) dst[i] = (__bf16)src[i];
}

// ---------------------------------------------------------------------------
// WMMA helper: D = A(16x32 bf16) x B(32x16 bf16) + C(16x16 f32)
// ---------------------------------------------------------------------------
__device__ __forceinline__ v8f wmma_bf16(v16bf a, v16bf b, v8f c) {
  return __builtin_amdgcn_wmma_f32_16x16x32_bf16(
      /*neg_a=*/false, a, /*neg_b=*/false, b,
      /*c_mod=*/(short)0, c, /*reuse_a=*/false, /*reuse_b=*/false);
}

// Load A operand (16x32 tile of X, row-major bf16 in LDS) in contiguous k order (B uses the same):
// lanes 0-15 hold row M=lr with K = [0..15];
// lanes 16-31 hold row M=lr with K = [16..31].
__device__ __forceinline__ v16bf load_a(const __bf16* __restrict__ aRow,
                                        int kb, int hv) {
  v8bf a0 = *(const v8bf*)(aRow + kb + hv * 16);
  v8bf a1 = *(const v8bf*)(aRow + kb + 8 + hv * 16);
  return __builtin_shufflevector(a0, a1, 0, 1, 2, 3, 4, 5, 6, 7,
                                 8, 9, 10, 11, 12, 13, 14, 15);
}

// ---------------------------------------------------------------------------
// Register-blocked GEMM stage: sOut(32x1024 bf16) = relu(sA @ Wg^T + bias).
// Each wave computes 2 row-tiles x 4 col-tiles per outer iteration:
// per k-step, 2 A loads + 4 B loads feed 8 WMMAs.
// Wave w owns column tiles ct = w + 8*(4*i + jj), i in {0,1}, jj in {0..3}.
// ---------------------------------------------------------------------------
__device__ __forceinline__ void gemm_relu(__bf16* __restrict__ sOut,
                                          const __bf16* __restrict__ sA,
                                          const __bf16* __restrict__ Wg,
                                          const float* __restrict__ bias,
                                          int w, int hv, int lr) {
  const __bf16* aRow0 = sA + lr * FD;
  const __bf16* aRow1 = sA + (16 + lr) * FD;
#pragma unroll 1
  for (int i = 0; i < 2; ++i) {
    int n[4];
    const __bf16* bRow[4];
    v8f c[2][4];
#pragma unroll
    for (int jj = 0; jj < 4; ++jj) {
      n[jj] = (w + 8 * (4 * i + jj)) * 16 + lr;
      const float bv = bias[n[jj]];
#pragma unroll
      for (int v = 0; v < 8; ++v) { c[0][jj][v] = bv; c[1][jj][v] = bv; }
      bRow[jj] = Wg + (long)n[jj] * FD + hv * 16;
    }
#pragma unroll 2
    for (int kb = 0; kb < FD; kb += 32) {
      v16bf a0 = load_a(aRow0, kb, hv);
      v16bf a1 = load_a(aRow1, kb, hv);
#pragma unroll
      for (int jj = 0; jj < 4; ++jj) {
        v16bf b = *(const v16bf*)(bRow[jj] + kb);
        c[0][jj] = wmma_bf16(a0, b, c[0][jj]);
        c[1][jj] = wmma_bf16(a1, b, c[1][jj]);
      }
    }
    // D layout: VGPR v -> row (v + 8*hv), col = lane&15
#pragma unroll
    for (int jj = 0; jj < 4; ++jj)
#pragma unroll
      for (int rt = 0; rt < 2; ++rt)
#pragma unroll
        for (int v = 0; v < 8; ++v) {
          const int r = rt * 16 + v + 8 * hv;
          sOut[r * FD + n[jj]] = (__bf16)fmaxf(c[rt][jj][v], 0.0f);
        }
  }
}

// ---------------------------------------------------------------------------
// Register-blocked fused stage: a = sA @ Wg^T + bias;
// eAcc[rt][v] += sD[r,n] * a[r,n] over this wave's column tiles.
// ---------------------------------------------------------------------------
__device__ __forceinline__ void gemm_dot(float eAcc[2][8],
                                         const __bf16* __restrict__ sA,
                                         const __bf16* __restrict__ sD,
                                         const __bf16* __restrict__ Wg,
                                         const float* __restrict__ bias,
                                         int w, int hv, int lr) {
  const __bf16* aRow0 = sA + lr * FD;
  const __bf16* aRow1 = sA + (16 + lr) * FD;
#pragma unroll 1
  for (int i = 0; i < 2; ++i) {
    int n[4];
    const __bf16* bRow[4];
    v8f c[2][4];
#pragma unroll
    for (int jj = 0; jj < 4; ++jj) {
      n[jj] = (w + 8 * (4 * i + jj)) * 16 + lr;
      const float bv = bias[n[jj]];
#pragma unroll
      for (int v = 0; v < 8; ++v) { c[0][jj][v] = bv; c[1][jj][v] = bv; }
      bRow[jj] = Wg + (long)n[jj] * FD + hv * 16;
    }
#pragma unroll 2
    for (int kb = 0; kb < FD; kb += 32) {
      v16bf a0 = load_a(aRow0, kb, hv);
      v16bf a1 = load_a(aRow1, kb, hv);
#pragma unroll
      for (int jj = 0; jj < 4; ++jj) {
        v16bf b = *(const v16bf*)(bRow[jj] + kb);
        c[0][jj] = wmma_bf16(a0, b, c[0][jj]);
        c[1][jj] = wmma_bf16(a1, b, c[1][jj]);
      }
    }
#pragma unroll
    for (int jj = 0; jj < 4; ++jj)
#pragma unroll
      for (int rt = 0; rt < 2; ++rt)
#pragma unroll
        for (int v = 0; v < 8; ++v) {
          const int r = rt * 16 + v + 8 * hv;
          eAcc[rt][v] += (float)sD[r * FD + n[jj]] * c[rt][jj][v];
        }
  }
}

// ---------------------------------------------------------------------------
// energies[row] = sum_j relu(Q@W0^T)[row,j] * ( relu(K@W1^T) @ Wa^T + ba )[row,j]
// One block = 32 rows, 256 threads = 8 wave32s, LDS ~193 KB (CDNA5 WGP).
// ---------------------------------------------------------------------------
__launch_bounds__(256)
__global__ void energies_kernel(const float* __restrict__ key,
                                const float* __restrict__ query,
                                const __bf16* __restrict__ W0b,
                                const __bf16* __restrict__ W1b,
                                const __bf16* __restrict__ Wab,
                                const float* __restrict__ b0,
                                const float* __restrict__ b1,
                                const float* __restrict__ ba,
                                float* __restrict__ energies) {
  __shared__ __bf16 sX[TM * FD];      // staged input tile (bf16)
  __shared__ __bf16 sT[TM * FD];      // q_enc = relu(key @ W1^T + b1)
  __shared__ __bf16 sE[TM * FD];      // k_enc = relu(query @ W0^T + b0)
  __shared__ float  sEpart[8][TM];    // per-wave energy partials

  const int tid  = threadIdx.x;
  const int wave = tid >> 5;
  const int lane = tid & 31;
  const int hv   = lane >> 4;   // lane half (K interleave select)
  const int lr   = lane & 15;   // row/col within tile
  const long row0 = (long)blockIdx.x * TM;

  // ---- stage KEY tile into LDS as bf16 ----
  {
    const float4* src = (const float4*)(key + row0 * FD);
    for (int j = tid; j < TM * FD / 4; j += 256) {
      float4 v = src[j];
      __bf16* p = &sX[4 * j];
      p[0] = (__bf16)v.x; p[1] = (__bf16)v.y;
      p[2] = (__bf16)v.z; p[3] = (__bf16)v.w;
    }
    // hint: pull the query tile toward L2 while we grind on key
    __builtin_prefetch(query + row0 * FD + (long)tid * (TM * FD / 256), 0, 1);
  }
  __syncthreads();

  // q_enc tile
  gemm_relu(sT, sX, W1b, b1, wave, hv, lr);
  __syncthreads();

  // ---- stage QUERY tile into LDS (reuse sX) ----
  {
    const float4* src = (const float4*)(query + row0 * FD);
    for (int j = tid; j < TM * FD / 4; j += 256) {
      float4 v = src[j];
      __bf16* p = &sX[4 * j];
      p[0] = (__bf16)v.x; p[1] = (__bf16)v.y;
      p[2] = (__bf16)v.z; p[3] = (__bf16)v.w;
    }
  }
  __syncthreads();

  // k_enc tile
  gemm_relu(sE, sX, W0b, b0, wave, hv, lr);
  __syncthreads();

  // attn GEMM fused with elementwise dot against k_enc
  float eAcc[2][8];
#pragma unroll
  for (int rt = 0; rt < 2; ++rt)
#pragma unroll
    for (int v = 0; v < 8; ++v) eAcc[rt][v] = 0.0f;
  gemm_dot(eAcc, sT, sE, Wab, ba, wave, hv, lr);

  // deterministic cross-lane reduce within each 16-lane half
#pragma unroll
  for (int rt = 0; rt < 2; ++rt)
#pragma unroll
    for (int v = 0; v < 8; ++v) {
      float val = eAcc[rt][v];
      val += __shfl_xor(val, 1, 32);
      val += __shfl_xor(val, 2, 32);
      val += __shfl_xor(val, 4, 32);
      val += __shfl_xor(val, 8, 32);
      if (lr == 0) sEpart[wave][rt * 16 + v + 8 * hv] = val;
    }
  __syncthreads();

  if (tid < TM) {
    float e = 0.0f;
#pragma unroll
    for (int w2 = 0; w2 < 8; ++w2) e += sEpart[w2][tid];
    energies[row0 + tid] = e;
  }
}

// ---------------------------------------------------------------------------
// Single-block softmax over 65536 energies -> alpha (deterministic trees)
// ---------------------------------------------------------------------------
__launch_bounds__(1024)
__global__ void softmax_kernel(const float* __restrict__ e,
                               float* __restrict__ alpha, int n) {
  __shared__ float red[1024];
  const int tid = threadIdx.x;
  float m = -3.4e38f;
  for (int i = tid; i < n; i += 1024) m = fmaxf(m, e[i]);
  red[tid] = m; __syncthreads();
  for (int s = 512; s > 0; s >>= 1) {
    if (tid < s) red[tid] = fmaxf(red[tid], red[tid + s]);
    __syncthreads();
  }
  const float mx = red[0];
  __syncthreads();
  float sum = 0.0f;
  for (int i = tid; i < n; i += 1024) sum += __expf(e[i] - mx);
  red[tid] = sum; __syncthreads();
  for (int s = 512; s > 0; s >>= 1) {
    if (tid < s) red[tid] += red[tid + s];
    __syncthreads();
  }
  const float inv = 1.0f / red[0];
  for (int i = tid; i < n; i += 1024) alpha[i] = __expf(e[i] - mx) * inv;
}

// ---------------------------------------------------------------------------
// context = alpha @ value, deterministic two-stage reduction (no atomics)
// ---------------------------------------------------------------------------
__launch_bounds__(256)
__global__ void ctx_partial_kernel(const float* __restrict__ alpha,
                                   const float* __restrict__ value,
                                   float* __restrict__ partial) {
  __shared__ float sA[NCH];
  const int tid = threadIdx.x;
  const int col = blockIdx.x * 256 + tid;
  const long n0 = (long)blockIdx.y * NCH;
  for (int i = tid; i < NCH; i += 256) sA[i] = alpha[n0 + i];
  __syncthreads();
  float acc = 0.0f;
#pragma unroll 4
  for (int j = 0; j < NCH; ++j)
    acc = fmaf(sA[j], value[(n0 + j) * FD + col], acc);
  partial[(long)blockIdx.y * FD + col] = acc;
}

__launch_bounds__(256)
__global__ void ctx_final_kernel(const float* __restrict__ partial,
                                 float* __restrict__ out, int nch) {
  const int col = blockIdx.x * 256 + threadIdx.x;
  float s = 0.0f;
  for (int ch = 0; ch < nch; ++ch) s += partial[(long)ch * FD + col];
  out[col] = s;
}

// ---------------------------------------------------------------------------
// Host-side launcher
// Inputs (setup_inputs order): key, query, value, W0, b0, W1, b1, Wa, ba
// ---------------------------------------------------------------------------
extern "C" void kernel_launch(void* const* d_in, const int* in_sizes, int n_in,
                              void* d_out, int out_size, void* d_ws,
                              size_t ws_size, hipStream_t stream) {
  const float* key   = (const float*)d_in[0];
  const float* query = (const float*)d_in[1];
  const float* value = (const float*)d_in[2];
  const float* W0    = (const float*)d_in[3];
  const float* b0    = (const float*)d_in[4];
  const float* W1    = (const float*)d_in[5];
  const float* b1    = (const float*)d_in[6];
  const float* Wa    = (const float*)d_in[7];
  const float* ba    = (const float*)d_in[8];
  float* out = (float*)d_out;

  // workspace layout (≈7 MB total)
  char* ws = (char*)d_ws;
  __bf16* W0b = (__bf16*)(ws);                                  // 2 MB
  __bf16* W1b = (__bf16*)(ws + ((size_t)2 << 20));              // 2 MB
  __bf16* Wab = (__bf16*)(ws + ((size_t)4 << 20));              // 2 MB
  float* energies = (float*)(ws + ((size_t)6 << 20));           // 256 KB
  float* alpha    = (float*)(ws + ((size_t)6 << 20) + (size_t)NARC * 4);
  float* partial  = (float*)(ws + ((size_t)6 << 20) + (size_t)NARC * 8);

  const int WN = FD * FD;
  cvt_f32_bf16<<<WN / 256, 256, 0, stream>>>(W0, W0b, WN);
  cvt_f32_bf16<<<WN / 256, 256, 0, stream>>>(W1, W1b, WN);
  cvt_f32_bf16<<<WN / 256, 256, 0, stream>>>(Wa, Wab, WN);

  energies_kernel<<<NARC / TM, 256, 0, stream>>>(
      key, query, W0b, W1b, Wab, b0, b1, ba, energies);

  softmax_kernel<<<1, 1024, 0, stream>>>(energies, alpha, NARC);

  ctx_partial_kernel<<<dim3(FD / 256, NARC / NCH), 256, 0, stream>>>(
      alpha, value, partial);
  ctx_final_kernel<<<FD / 256, 256, 0, stream>>>(partial, out, NARC / NCH);
}
